// DensitySetAbstraction_78726750536382
// MI455X (gfx1250) — matrix operand order repaired and checked by reading, not compiled, not run on hardware
//
#include <hip/hip_runtime.h>
#include <hip/hip_bf16.h>

#define B_    16
#define N_    4096
#define CF_   64
#define S_    1024
#define NS_   32
#define INCH_ 67
#define KPAD0 96
#define C2_   128
#define P_    (B_ * S_ * NS_)   /* 524288 positions */
#define EPS_  1e-5f

typedef __bf16 bf16_t;
typedef __attribute__((ext_vector_type(16))) __bf16 v16bf;
typedef __attribute__((ext_vector_type(8)))  float  v8f;
typedef unsigned long long u64;
typedef unsigned int u32;

union Frag16 { v16bf v; unsigned int u[8]; };

static __device__ __forceinline__ u64 shfl_xor_u64(u64 x, int mask) {
  u32 lo = (u32)x, hi = (u32)(x >> 32);
  lo = __shfl_xor(lo, mask, 32);
  hi = __shfl_xor(hi, mask, 32);
  return ((u64)hi << 32) | lo;
}

// ---------------------------------------------------------------------------
// 1) FPS: one block per batch. Points in regs + LDS (fast center fetch).
//    Per iteration: reg dist update, wave shuffle arg-max on packed 64-bit
//    keys, ONE barrier via double-buffered candidate stage.
// ---------------------------------------------------------------------------
__global__ __launch_bounds__(256) void fps_kernel(
    const float* __restrict__ xyz, const float* __restrict__ density,
    int* __restrict__ fps_idx, float* __restrict__ new_xyz,
    float* __restrict__ dcen) {
  const int b = blockIdx.x;
  const int t = threadIdx.x;
  const int lane = t & 31, wid = t >> 5;
  const float* xb = xyz + (size_t)b * N_ * 3;

  __shared__ float sx[N_], sy[N_], sz[N_];
  __shared__ int   sfar[S_];
  __shared__ u64   wkey[2][8];

  float px[16], py[16], pz[16], dist[16];
#pragma unroll
  for (int j = 0; j < 16; ++j) {
    const int i = t + 256 * j;
    px[j] = xb[i * 3 + 0]; py[j] = xb[i * 3 + 1]; pz[j] = xb[i * 3 + 2];
    sx[i] = px[j]; sy[i] = py[j]; sz[i] = pz[j];
    dist[j] = 1e10f;
  }
  __syncthreads();

  int far = 0;
  for (int it = 0; it < S_; ++it) {
    if (t == 0) sfar[it] = far;               // record incoming 'far'
    const float cx = sx[far], cy = sy[far], cz = sz[far];
    float bv = -1.0f; int bi = 0;
#pragma unroll
    for (int j = 0; j < 16; ++j) {
      const float dx = px[j] - cx, dy = py[j] - cy, dz = pz[j] - cz;
      const float d = dx * dx + dy * dy + dz * dz;
      dist[j] = fminf(dist[j], d);
      if (dist[j] > bv) { bv = dist[j]; bi = t + 256 * j; }
    }
    // pack: dist bits high, complemented idx low -> max picks smallest idx tie
    u64 key = ((u64)__float_as_uint(bv) << 32) | (u32)(N_ - 1 - bi);
#pragma unroll
    for (int msk = 16; msk > 0; msk >>= 1) {
      const u64 o = shfl_xor_u64(key, msk);
      key = (o > key) ? o : key;
    }
    if (lane == 0) wkey[it & 1][wid] = key;
    __syncthreads();
    u64 best = wkey[it & 1][0];
#pragma unroll
    for (int w = 1; w < 8; ++w) { const u64 o = wkey[it & 1][w]; best = (o > best) ? o : best; }
    far = N_ - 1 - (int)(best & 0xffffffffu);
  }
  __syncthreads();
  for (int i = t; i < S_; i += 256) {
    const int f = sfar[i];
    fps_idx[b * S_ + i] = f;
    new_xyz[(b * S_ + i) * 3 + 0] = sx[f];
    new_xyz[(b * S_ + i) * 3 + 1] = sy[f];
    new_xyz[(b * S_ + i) * 3 + 2] = sz[f];
    dcen[b * S_ + i] = density[(size_t)b * N_ + f];
  }
}

// ---------------------------------------------------------------------------
// 2) Density batch statistics over (B,S) = 16384 values. Single block.
// ---------------------------------------------------------------------------
__global__ __launch_bounds__(256) void density_stats_kernel(
    const float* __restrict__ dcen, float* __restrict__ dstats) {
  __shared__ float s1[256], s2[256];
  const int t = threadIdx.x;
  float a = 0.f, q = 0.f;
  for (int i = t; i < B_ * S_; i += 256) { const float v = dcen[i]; a += v; q += v * v; }
  s1[t] = a; s2[t] = q;
  __syncthreads();
  for (int off = 128; off > 0; off >>= 1) {
    if (t < off) { s1[t] += s1[t + off]; s2[t] += s2[t + off]; }
    __syncthreads();
  }
  if (t == 0) { dstats[0] = s1[0]; dstats[1] = s2[0]; }
}

// ---------------------------------------------------------------------------
// 3) kNN: one block per (b,s). Register-resident distances, packed-min wave
//    shuffle reduction, one barrier per selection (double-buffered stage).
// ---------------------------------------------------------------------------
__global__ __launch_bounds__(256) void knn_kernel(
    const float* __restrict__ xyz, const float* __restrict__ new_xyz,
    int* __restrict__ knn_idx) {
  const int blk = blockIdx.x;
  const int b = blk >> 10;                 // S_ = 1024
  const int s = blk & (S_ - 1);
  const int t = threadIdx.x;
  const int lane = t & 31, wid = t >> 5;

  __shared__ u64 wkey[2][8];
  __shared__ int chosen[NS_];

  const float* xb = xyz + (size_t)b * N_ * 3;
  const float cx = new_xyz[(b * S_ + s) * 3 + 0];
  const float cy = new_xyz[(b * S_ + s) * 3 + 1];
  const float cz = new_xyz[(b * S_ + s) * 3 + 2];

  float dreg[16];
#pragma unroll
  for (int j = 0; j < 16; ++j) {
    const int i = t + 256 * j;
    const float dx = xb[i * 3 + 0] - cx;
    const float dy = xb[i * 3 + 1] - cy;
    const float dz = xb[i * 3 + 2] - cz;
    dreg[j] = dx * dx + dy * dy + dz * dz;
  }

  for (int sel = 0; sel < NS_; ++sel) {
    float bv = 1e30f; int bi = 0;
#pragma unroll
    for (int j = 0; j < 16; ++j) {
      if (dreg[j] < bv) { bv = dreg[j]; bi = t + 256 * j; }
    }
    u64 key = ((u64)__float_as_uint(bv) << 32) | (u32)bi;  // min tie -> lowest idx
#pragma unroll
    for (int msk = 16; msk > 0; msk >>= 1) {
      const u64 o = shfl_xor_u64(key, msk);
      key = (o < key) ? o : key;
    }
    if (lane == 0) wkey[sel & 1][wid] = key;
    __syncthreads();
    u64 best = wkey[sel & 1][0];
#pragma unroll
    for (int w = 1; w < 8; ++w) { const u64 o = wkey[sel & 1][w]; best = (o < best) ? o : best; }
    const int far = (int)(best & 0xffffffffu);
    if (t == 0) chosen[sel] = far;
#pragma unroll
    for (int j = 0; j < 16; ++j) {                 // owner invalidates its copy
      if (far == t + 256 * j) dreg[j] = 1e30f;
    }
  }
  __syncthreads();
  if (t < NS_) knn_idx[(size_t)blk * NS_ + t] = chosen[t];
}

// ---------------------------------------------------------------------------
// 4) Weight conversion: w[cout,cin] f32 -> wt[cout,kpad] bf16 (K contiguous).
// ---------------------------------------------------------------------------
__global__ __launch_bounds__(256) void wconv_kernel(
    const float* __restrict__ w, bf16_t* __restrict__ wt,
    int cout, int cin, int kpad) {
  const int i = blockIdx.x * 256 + threadIdx.x;
  if (i >= cout * kpad) return;
  const int n = i / kpad, k = i - n * kpad;
  wt[i] = (k < cin) ? (bf16_t)w[n * cin + k] : (bf16_t)0.0f;
}

// ---------------------------------------------------------------------------
// 5) Gather features: Xg[p, 0:3] = xyz[idx]-center, [3:67] = points, [67:96]=0
// ---------------------------------------------------------------------------
__global__ void gather_kernel(const float* __restrict__ xyz,
                              const float* __restrict__ points,
                              const float* __restrict__ new_xyz,
                              const int* __restrict__ knn_idx,
                              bf16_t* __restrict__ Xg) {
  const int c = threadIdx.x;                        // 0..95
  const int p = blockIdx.x * blockDim.y + threadIdx.y;
  const int b = p >> 15;                            // S_*NS_ = 32768
  const int s = (p & 32767) >> 5;
  const int idx = knn_idx[p];
  float v;
  if (c < 3)        v = xyz[((size_t)b * N_ + idx) * 3 + c] - new_xyz[(b * S_ + s) * 3 + c];
  else if (c < 67)  v = points[((size_t)b * CF_ + (c - 3)) * N_ + idx];
  else              v = 0.0f;
  Xg[(size_t)p * KPAD0 + c] = (bf16_t)v;
}

// ---------------------------------------------------------------------------
// 6) bf16 WMMA GEMM, N-fused: each wave computes 16 rows x NT*16 cols,
//    loading the A fragment once per K-step. Per-channel BN sums fused into
//    the epilogue (LDS ds_add_f32, then one global atomic per channel/block).
//    Fragment layouts per CDNA5 ISA 7.12.2.
// ---------------------------------------------------------------------------
template<int NT>
__global__ __launch_bounds__(256) void gemm_bf16_kernel(
    const bf16_t* __restrict__ X, const bf16_t* __restrict__ Wt,
    bf16_t* __restrict__ Y, int Kpad, float* __restrict__ gstats) {
  const int C = NT * 16;
  __shared__ float s1s[128], s2s[128];
  const int t = threadIdx.x;
  if (t < C) { s1s[t] = 0.f; s2s[t] = 0.f; }
  __syncthreads();

  const int wid = t >> 5, lane = t & 31;
  const int mBase = (blockIdx.x * 8 + wid) * 16;
  const int m = lane & 15, half = lane >> 4;

  const bf16_t* xrow = X + (size_t)(mBase + m) * Kpad;

  v8f zero = {0.f, 0.f, 0.f, 0.f, 0.f, 0.f, 0.f, 0.f};
  v8f acc[NT];
#pragma unroll
  for (int nt = 0; nt < NT; ++nt) acc[nt] = zero;

  for (int kk = 0; kk < Kpad; kk += 32) {
    Frag16 a;
#pragma unroll
    for (int j = 0; j < 8; ++j) {                  // A: 16x32 bf16 pairs
      const int k0 = kk + ((j >> 2) << 4) + (half << 3) + ((j & 3) << 1);
      a.u[j] = *(const unsigned int*)(xrow + k0);
    }
#pragma unroll
    for (int nt = 0; nt < NT; ++nt) {
      Frag16 bb;
      const bf16_t* wrow = Wt + (size_t)(nt * 16 + m) * Kpad;
#pragma unroll
      for (int j = 0; j < 8; ++j) {                // B: 32x16, lanes0-15 K=0..7,16..23
        const int k0 = kk + ((j >> 2) << 4) + (half << 3) + ((j & 3) << 1);
        bb.u[j] = *(const unsigned int*)(wrow + k0);
      }
      acc[nt] = __builtin_amdgcn_wmma_f32_16x16x32_bf16(
          false, a.v, false, bb.v, (short)0, acc[nt], false, false);
    }
  }

#pragma unroll
  for (int nt = 0; nt < NT; ++nt) {
    const int col = nt * 16 + m;
    float ls1 = 0.f, ls2 = 0.f;
#pragma unroll
    for (int r = 0; r < 8; ++r) {                  // D: VGPR r -> M = r + 8*half
      const float v = acc[nt][r];
      const int row = mBase + r + (half << 3);
      Y[(size_t)row * C + col] = (bf16_t)v;
      ls1 += v; ls2 += v * v;
    }
    atomicAdd(&s1s[col], ls1);
    atomicAdd(&s2s[col], ls2);
  }
  __syncthreads();
  if (t < C) {
    atomicAdd(&gstats[t], s1s[t]);
    atomicAdd(&gstats[C + t], s2s[t]);
  }
}

// ---------------------------------------------------------------------------
// 7) In-place BN (batch stats) + ReLU on bf16 activations.
// ---------------------------------------------------------------------------
__global__ __launch_bounds__(256) void bnrelu_kernel(
    bf16_t* __restrict__ Y, const float* __restrict__ stats,
    const float* __restrict__ g, const float* __restrict__ be, int C) {
  const size_t idx = (size_t)blockIdx.x * 256 + threadIdx.x;
  if (idx >= (size_t)P_ * C) return;
  const int c = (int)(idx & (size_t)(C - 1));
  const float invP = 1.0f / (float)P_;
  const float mean = stats[c] * invP;
  const float var  = stats[C + c] * invP - mean * mean;
  const float sc   = rsqrtf(var + EPS_) * g[c];
  float v = (float)Y[idx];
  v = (v - mean) * sc + be[c];
  Y[idx] = (bf16_t)fmaxf(v, 0.0f);
}

// ---------------------------------------------------------------------------
// 8) Final: BN2+ReLU on layer-2 output, density weight folded, max over 32
//    samples -> out_feat[B,128,S].
// ---------------------------------------------------------------------------
__global__ __launch_bounds__(256) void final_kernel(
    const bf16_t* __restrict__ Y2, const float* __restrict__ stats2,
    const float* __restrict__ g2, const float* __restrict__ be2,
    const float* __restrict__ dcen, const float* __restrict__ dstats,
    const float* __restrict__ wd, const float* __restrict__ gd,
    const float* __restrict__ bed, float* __restrict__ out) {
  const int t = blockIdx.x * 256 + threadIdx.x;
  if (t >= B_ * C2_ * S_) return;
  const int s = t & (S_ - 1);
  const int c = (t >> 10) & (C2_ - 1);
  const int b = t >> 17;

  const float invP = 1.0f / (float)P_;
  const float mean = stats2[c] * invP;
  const float var  = stats2[C2_ + c] * invP - mean * mean;
  const float sc   = rsqrtf(var + EPS_) * g2[c];
  const float beta = be2[c];

  const float invBS = 1.0f / (float)(B_ * S_);
  const float dmean = dstats[0] * invBS;
  const float dvar  = dstats[1] * invBS - dmean * dmean;
  const float w  = wd[c];
  const float dc = dcen[b * S_ + s];
  float dw = (w * dc - w * dmean) * rsqrtf(w * w * dvar + EPS_) * gd[c] + bed[c];
  dw = fmaxf(dw, 0.0f);

  const bf16_t* yp = Y2 + (size_t)((b * S_ + s) * NS_) * C2_ + c;
  float m = 0.0f;
#pragma unroll
  for (int k = 0; k < NS_; ++k) {
    const float y = (float)yp[(size_t)k * C2_];
    const float v = fmaxf((y - mean) * sc + beta, 0.0f) * dw;
    m = fmaxf(m, v);
  }
  out[t] = m;    // t == ((b*128)+c)*1024 + s
}

// ---------------------------------------------------------------------------
extern "C" void kernel_launch(void* const* d_in, const int* in_sizes, int n_in,
                              void* d_out, int out_size, void* d_ws, size_t ws_size,
                              hipStream_t stream) {
  (void)in_sizes; (void)n_in; (void)out_size; (void)ws_size;
  const float* xyz     = (const float*)d_in[0];
  const float* points  = (const float*)d_in[1];
  const float* density = (const float*)d_in[2];
  const float* w0  = (const float*)d_in[3];
  const float* g0  = (const float*)d_in[5];
  const float* be0 = (const float*)d_in[6];
  const float* w1  = (const float*)d_in[7];
  const float* g1  = (const float*)d_in[9];
  const float* be1 = (const float*)d_in[10];
  const float* w2  = (const float*)d_in[11];
  const float* g2  = (const float*)d_in[13];
  const float* be2 = (const float*)d_in[14];
  const float* wd  = (const float*)d_in[15];
  const float* gd  = (const float*)d_in[17];
  const float* bed = (const float*)d_in[18];

  float* out_xyz  = (float*)d_out;              // (B,S,3)
  float* out_feat = out_xyz + B_ * S_ * 3;      // (B,128,S)

  char* ws = (char*)d_ws;
  size_t off = 0;
  auto carve = [&](size_t bytes) -> void* {
    void* p = (void*)(ws + off);
    off += (bytes + 255) & ~(size_t)255;
    return p;
  };
  int*    fps_idx = (int*)   carve((size_t)B_ * S_ * sizeof(int));
  int*    knn_idx = (int*)   carve((size_t)B_ * S_ * NS_ * sizeof(int));
  float*  dcen    = (float*) carve((size_t)B_ * S_ * sizeof(float));
  float*  dstats  = (float*) carve(2 * sizeof(float));
  float*  stats   = (float*) carve(768 * sizeof(float)); // 3 x 256 floats
  bf16_t* w0t     = (bf16_t*)carve((size_t)64  * KPAD0 * sizeof(bf16_t));
  bf16_t* w1t     = (bf16_t*)carve((size_t)64  * 64    * sizeof(bf16_t));
  bf16_t* w2t     = (bf16_t*)carve((size_t)128 * 64    * sizeof(bf16_t));
  bf16_t* Xg      = (bf16_t*)carve((size_t)P_ * KPAD0 * sizeof(bf16_t)); // also Y1
  bf16_t* Y0      = (bf16_t*)carve((size_t)P_ * 64    * sizeof(bf16_t));
  bf16_t* Y2      = (bf16_t*)carve((size_t)P_ * 128   * sizeof(bf16_t));
  bf16_t* Y1      = Xg;   // Xg dead after layer-0 GEMM

  float* stats0 = stats;
  float* stats1 = stats + 256;
  float* stats2 = stats + 512;

  hipMemsetAsync(stats, 0, 768 * sizeof(float), stream);

  fps_kernel<<<B_, 256, 0, stream>>>(xyz, density, fps_idx, out_xyz, dcen);
  density_stats_kernel<<<1, 256, 0, stream>>>(dcen, dstats);
  knn_kernel<<<B_ * S_, 256, 0, stream>>>(xyz, out_xyz, knn_idx);

  wconv_kernel<<<(64  * KPAD0 + 255) / 256, 256, 0, stream>>>(w0, w0t, 64,  INCH_, KPAD0);
  wconv_kernel<<<(64  * 64    + 255) / 256, 256, 0, stream>>>(w1, w1t, 64,  64,    64);
  wconv_kernel<<<(128 * 64    + 255) / 256, 256, 0, stream>>>(w2, w2t, 128, 64,    64);

  {
    dim3 blk(96, 4);
    gather_kernel<<<P_ / 4, blk, 0, stream>>>(xyz, points, out_xyz, knn_idx, Xg);
  }

  const int gemmBlocks = (P_ / 16) / 8;   // 4096 blocks, 8 waves each

  // Layer 0: [P,96] -> [P,64], stats fused
  gemm_bf16_kernel<4><<<gemmBlocks, 256, 0, stream>>>(Xg, w0t, Y0, KPAD0, stats0);
  bnrelu_kernel<<<((size_t)P_ * 64 + 255) / 256, 256, 0, stream>>>(Y0, stats0, g0, be0, 64);

  // Layer 1: [P,64] -> [P,64] (into Xg region)
  gemm_bf16_kernel<4><<<gemmBlocks, 256, 0, stream>>>(Y0, w1t, Y1, 64, stats1);
  bnrelu_kernel<<<((size_t)P_ * 64 + 255) / 256, 256, 0, stream>>>(Y1, stats1, g1, be1, 64);

  // Layer 2: [P,64] -> [P,128]
  gemm_bf16_kernel<8><<<gemmBlocks, 256, 0, stream>>>(Y1, w2t, Y2, 64, stats2);

  // Final fused BN2 + density weighting + max over samples
  final_kernel<<<(B_ * C2_ * S_ + 255) / 256, 256, 0, stream>>>(
      Y2, stats2, g2, be2, dcen, dstats, wd, gd, bed, out_feat);
}
